// MambaBlock_19619410608648
// MI455X (gfx1250) — matrix operand order repaired and checked by reading, not compiled, not run on hardware
//
#include <hip/hip_runtime.h>
#include <cstdint>

#define DMODEL 1024
#define DINNER 2048
#define DSTATE 16
#define DCONV  4
#define DTRANK 64
#define NXPROJ 96          // DTRANK + 2*DSTATE
#define BATCH  2
#define SEQ    4096
#define MTOK   (BATCH*SEQ) // 8192 tokens
#define LNEPS  1e-5f

typedef __attribute__((ext_vector_type(16))) __bf16 v16bf;
typedef __attribute__((ext_vector_type(8)))  float  v8f;

union FragAB { uint4 q[2]; v16bf v; };

__device__ __forceinline__ unsigned short f2bf(float f) {
    unsigned u = __float_as_uint(f);
    u += 0x7fffu + ((u >> 16) & 1u);          // round-to-nearest-even
    return (unsigned short)(u >> 16);
}
__device__ __forceinline__ float silu(float x) { return x / (1.f + __expf(-x)); }

// CDNA5 async memory->LDS copy (VGLOBAL GLOBAL_LOAD_ASYNC_TO_LDS_B128, ASYNCcnt)
__device__ __forceinline__ void async_copy_b128(unsigned lds_off, const void* gaddr) {
    asm volatile("global_load_async_to_lds_b128 %0, %1, off"
                 :: "v"(lds_off), "v"((unsigned long long)(size_t)gaddr)
                 : "memory");
}
__device__ __forceinline__ void wait_async_le0() {
    asm volatile("s_wait_asynccnt 0" ::: "memory");
}
__device__ __forceinline__ void wait_async_le3() {   // keep newest stage in flight
    asm volatile("s_wait_asynccnt 3" ::: "memory");
}
__device__ __forceinline__ unsigned lds_off32(const void* p) {
    return (unsigned)(size_t)p;   // generic LDS addr low 32 bits = DS-space offset
}

// ------------------------------------- f32 [K][N] -> bf16 [N][K] (transpose)
// K, N multiples of 32. Tiled through LDS so both sides stay coalesced.
__global__ void cvt_T_bf16(const float* __restrict__ s, unsigned short* __restrict__ d,
                           int K, int N) {
    __shared__ float tile[32][33];
    int kb = blockIdx.y * 32, nb = blockIdx.x * 32;
    int tx = threadIdx.x & 31, ty = threadIdx.x >> 5;   // ty: 0..7
    #pragma unroll
    for (int i = 0; i < 4; ++i)
        tile[ty + i * 8][tx] = s[(size_t)(kb + ty + i * 8) * N + nb + tx];
    __syncthreads();
    #pragma unroll
    for (int i = 0; i < 4; ++i)
        d[(size_t)(nb + ty + i * 8) * K + kb + tx] = f2bf(tile[tx][ty + i * 8]);
}

// ---------------------------------------------------------------- LayerNorm
__global__ void ln_kernel(const float* __restrict__ x, const float* __restrict__ g,
                          const float* __restrict__ b, unsigned short* __restrict__ xn) {
    __shared__ float2 red[256];
    int row = blockIdx.x;
    const float* xr = x + (size_t)row * DMODEL;
    float v[4]; float s = 0.f, s2 = 0.f;
    #pragma unroll
    for (int j = 0; j < 4; ++j) {
        v[j] = xr[threadIdx.x + j * 256];
        s += v[j]; s2 += v[j] * v[j];
    }
    red[threadIdx.x] = make_float2(s, s2);
    __syncthreads();
    for (int off = 128; off > 0; off >>= 1) {
        if (threadIdx.x < off) {
            float2 o = red[threadIdx.x + off];
            red[threadIdx.x].x += o.x; red[threadIdx.x].y += o.y;
        }
        __syncthreads();
    }
    float mu  = red[0].x * (1.f / DMODEL);
    float var = red[0].y * (1.f / DMODEL) - mu * mu;
    float rs  = __frsqrt_rn(var + LNEPS);
    unsigned short* o = xn + (size_t)row * DMODEL;
    #pragma unroll
    for (int j = 0; j < 4; ++j) {
        int c = threadIdx.x + j * 256;
        o[c] = f2bf((v[j] - mu) * rs * g[c] + b[c]);
    }
}

// ------------------------------------------------- generic bf16 WMMA GEMM
// C[M,N](f32) = A[M,K](bf16, row-major) @ Bt[N,K](bf16, PRE-TRANSPOSED)
// (+ addend[M,N] if non-null). block = 256 threads (8 waves),
// tile 128(M) x 64(N), K step 32. Double-buffered async-to-LDS staging:
// stage s+1 is issued before computing stage s; s_wait_asynccnt 3 retires
// only the oldest stage (async loads complete in order).
__global__ __launch_bounds__(256, 2)
void gemm_bf16_wmma(const unsigned short* __restrict__ A,
                    const unsigned short* __restrict__ Bt,
                    float* __restrict__ C,
                    const float* __restrict__ addend,
                    int M, int N, int K) {
    __shared__ unsigned short sa[2][128 * 40];  // A tiles [m][k], padded stride 40
    __shared__ unsigned short sb[2][64 * 40];   // B tiles [n][k], padded stride 40
    __shared__ uint4 dump[256];                 // dest for OOB-redirected copies
    const int t    = threadIdx.x;
    const int lane = t & 31;
    const int wid  = t >> 5;
    const int wm   = wid >> 1;                  // 0..3  (M direction)
    const int wn   = wid & 1;                   // 0..1  (N direction)
    const int m0   = blockIdx.y * 128;
    const int n0   = blockIdx.x * 64;

    const v8f vzero = {0.f,0.f,0.f,0.f,0.f,0.f,0.f,0.f};
    v8f acc[2][2];
    acc[0][0] = vzero; acc[0][1] = vzero; acc[1][0] = vzero; acc[1][1] = vzero;

    const int lr  = lane & 15;
    const int hi  = lane >> 4;                  // 0 or 1
    const int akb = hi * 8;                     // A lane K base (ISA 16-bit A layout)
    const int bkb = hi * 8;                     // B lane K base (ISA 16-bit B layout)

    // per-thread staging coordinates
    const int ar0 = t >> 2,        ac0 = t & 3;         // A chunk 0
    const int ar1 = (t + 256) >> 2, ac1 = (t + 256) & 3; // A chunk 1
    const int bn  = t >> 2,        bkc = t & 3;         // B row n (0..63)
    const bool b_ok = (n0 + bn) < N;
    const int  bn_src = b_ok ? (n0 + bn) : (N - 1);     // clamped (valid) source row

    // zero-fill OOB B rows of both buffers once; they are never overwritten
    if (!b_ok) {
        const uint4 z4 = make_uint4(0, 0, 0, 0);
        *reinterpret_cast<uint4*>(&sb[0][bn * 40 + bkc * 8]) = z4;
        *reinterpret_cast<uint4*>(&sb[1][bn * 40 + bkc * 8]) = z4;
    }

    const unsigned sa_off0[2] = { lds_off32(&sa[0][ar0 * 40 + ac0 * 8]),
                                  lds_off32(&sa[1][ar0 * 40 + ac0 * 8]) };
    const unsigned sa_off1[2] = { lds_off32(&sa[0][ar1 * 40 + ac1 * 8]),
                                  lds_off32(&sa[1][ar1 * 40 + ac1 * 8]) };
    const unsigned dump_off   = lds_off32(&dump[t]);
    const unsigned sb_off[2]  = { b_ok ? lds_off32(&sb[0][bn * 40 + bkc * 8]) : dump_off,
                                  b_ok ? lds_off32(&sb[1][bn * 40 + bkc * 8]) : dump_off };

    const unsigned short* aRow0 = A  + (size_t)(m0 + ar0) * K + ac0 * 8;
    const unsigned short* aRow1 = A  + (size_t)(m0 + ar1) * K + ac1 * 8;
    const unsigned short* bRow  = Bt + (size_t)bn_src     * K + bkc * 8;

    const int nk = K >> 5;
    // prologue: stage 0 into buffer 0 (exactly 3 asyncs per thread per stage)
    async_copy_b128(sa_off0[0], aRow0);
    async_copy_b128(sa_off1[0], aRow1);
    async_copy_b128(sb_off[0],  bRow);

    for (int s = 0; s < nk; ++s) {
        const int buf  = s & 1;
        const bool more = (s + 1) < nk;
        if (more) {
            const int k1 = (s + 1) << 5;
            async_copy_b128(sa_off0[buf ^ 1], aRow0 + k1);
            async_copy_b128(sa_off1[buf ^ 1], aRow1 + k1);
            async_copy_b128(sb_off[buf ^ 1],  bRow  + k1);
            if (s + 2 < nk)      // prefetch 2 stages ahead (global_prefetch_b8)
                __builtin_prefetch(aRow0 + ((s + 2) << 5), 0, 1);
        }
        if (more) wait_async_le3(); else wait_async_le0();
        __syncthreads();   // stage s fully in LDS (all waves)

        // ---- assemble fragments (ds_load_b128 pairs) + 4 WMMAs ----
        FragAB afr[2], bfr[2];
        #pragma unroll
        for (int mt = 0; mt < 2; ++mt) {
            int r = wm * 32 + mt * 16 + lr;
            afr[mt].q[0] = *reinterpret_cast<const uint4*>(&sa[buf][r * 40 + akb]);
            afr[mt].q[1] = *reinterpret_cast<const uint4*>(&sa[buf][r * 40 + akb + 16]);
        }
        #pragma unroll
        for (int nt = 0; nt < 2; ++nt) {
            int n = wn * 32 + nt * 16 + lr;
            bfr[nt].q[0] = *reinterpret_cast<const uint4*>(&sb[buf][n * 40 + bkb]);
            bfr[nt].q[1] = *reinterpret_cast<const uint4*>(&sb[buf][n * 40 + bkb + 16]);
        }
        #pragma unroll
        for (int mt = 0; mt < 2; ++mt)
            #pragma unroll
            for (int nt = 0; nt < 2; ++nt)
                acc[mt][nt] = __builtin_amdgcn_wmma_f32_16x16x32_bf16(
                    false, afr[mt].v, false, bfr[nt].v,
                    (short)0, acc[mt][nt], false, false);
        __syncthreads();   // all reads of buf done before it is refilled
    }

    // ---- store C (ISA C/D layout: VGPR r -> M=r (+8 for high lanes), N=lane&15)
    #pragma unroll
    for (int mt = 0; mt < 2; ++mt) {
        #pragma unroll
        for (int nt = 0; nt < 2; ++nt) {
            int gn = n0 + wn * 32 + nt * 16 + lr;
            if (gn >= N) continue;
            #pragma unroll
            for (int r = 0; r < 8; ++r) {
                int gm = m0 + wm * 32 + mt * 16 + r + hi * 8;
                size_t idx = (size_t)gm * N + gn;
                float val = acc[mt][nt][r];
                if (addend) val += addend[idx];
                C[idx] = val;
            }
        }
    }
}

// ------------------------------------------ depthwise causal conv + SiLU
__global__ void conv_silu_kernel(const float* __restrict__ xz,
                                 const float* __restrict__ cw,
                                 const float* __restrict__ cb,
                                 float* __restrict__ usil,
                                 unsigned short* __restrict__ ubf) {
    int idx = blockIdx.x * blockDim.x + threadIdx.x;
    if (idx >= MTOK * DINNER) return;
    int d  = idx % DINNER;
    int ml = idx / DINNER;
    int l  = ml % SEQ;
    int b  = ml / SEQ;
    float acc = cb[d];
    #pragma unroll
    for (int k = 0; k < DCONV; ++k) {
        int ls = l - (DCONV - 1) + k;
        if (ls >= 0)
            acc += xz[((size_t)(b * SEQ + ls)) * (2 * DINNER) + d] * cw[d * DCONV + k];
    }
    float s = silu(acc);
    usil[idx] = s;
    ubf[idx]  = f2bf(s);
}

// -------------------------------------- slice xdbc[:, :64] -> bf16 dt input
__global__ void slice_dt_kernel(const float* __restrict__ xdbc,
                                unsigned short* __restrict__ dtin) {
    int i = blockIdx.x * blockDim.x + threadIdx.x;
    if (i >= MTOK * DTRANK) return;
    int m = i / DTRANK, j = i % DTRANK;
    dtin[i] = f2bf(xdbc[(size_t)m * NXPROJ + j]);
}

// ------------------------------------------------- softplus(dt + bias)
__global__ void softplus_kernel(float* __restrict__ dt, const float* __restrict__ bias) {
    int i = blockIdx.x * blockDim.x + threadIdx.x;
    if (i >= MTOK * DINNER) return;
    float v = dt[i] + bias[i % DINNER];
    dt[i] = (v > 20.f) ? v : __logf(1.f + __expf(v));
}

// ----------------------------------- selective scan, one thread per (b,d)
__global__ void scan_kernel(const float* __restrict__ dt,
                            const float* __restrict__ usil,
                            const float* __restrict__ xdbc,
                            const float* __restrict__ xz,
                            const float* __restrict__ A_log,
                            const float* __restrict__ Dp,
                            unsigned short* __restrict__ ybf) {
    int tid = blockIdx.x * blockDim.x + threadIdx.x;
    if (tid >= BATCH * DINNER) return;
    int d = tid % DINNER;
    int b = tid / DINNER;
    float Arow[DSTATE];
    #pragma unroll
    for (int n = 0; n < DSTATE; ++n) Arow[n] = -__expf(A_log[d * DSTATE + n]);
    float Dv = Dp[d];
    float h[DSTATE];
    #pragma unroll
    for (int n = 0; n < DSTATE; ++n) h[n] = 0.f;

    for (int tstep = 0; tstep < SEQ; ++tstep) {
        size_t tok = (size_t)b * SEQ + tstep;
        float dtv = dt[tok * DINNER + d];
        float uv  = usil[tok * DINNER + d];
        const float* xrow = xdbc + tok * NXPROJ;
        float dtu = dtv * uv;
        float y = 0.f;
        #pragma unroll
        for (int n = 0; n < DSTATE; ++n) {
            float Bn = xrow[DTRANK + n];
            float Cn = xrow[DTRANK + DSTATE + n];
            h[n] = h[n] * __expf(dtv * Arow[n]) + dtu * Bn;
            y += h[n] * Cn;
        }
        float z   = xz[tok * (2 * DINNER) + DINNER + d];
        float out = (y + uv * Dv) * silu(z);
        ybf[tok * DINNER + d] = f2bf(out);
    }
}

// ------------------------------------------------------------------ launch
extern "C" void kernel_launch(void* const* d_in, const int* in_sizes, int n_in,
                              void* d_out, int out_size, void* d_ws, size_t ws_size,
                              hipStream_t stream) {
    const float* x     = (const float*)d_in[0];
    const float* lng   = (const float*)d_in[1];
    const float* lnb   = (const float*)d_in[2];
    const float* w_in  = (const float*)d_in[3];
    const float* cw    = (const float*)d_in[4];
    const float* cb    = (const float*)d_in[5];
    const float* w_xp  = (const float*)d_in[6];
    const float* w_dt  = (const float*)d_in[7];
    const float* b_dt  = (const float*)d_in[8];
    const float* Alog  = (const float*)d_in[9];
    const float* Dp    = (const float*)d_in[10];
    const float* w_out = (const float*)d_in[11];
    float* out = (float*)d_out;
    (void)in_sizes; (void)n_in; (void)out_size; (void)ws_size;

    char* ws = (char*)d_ws;
    size_t off = 0;
    auto alloc = [&](size_t bytes) -> void* {
        void* p = ws + off;
        off = (off + bytes + 255) & ~(size_t)255;
        return p;
    };
    unsigned short* xnbf  = (unsigned short*)alloc((size_t)MTOK * DMODEL * 2);
    unsigned short* w1T   = (unsigned short*)alloc((size_t)DMODEL * 2 * DINNER * 2);
    float*          xzf   = (float*)         alloc((size_t)MTOK * 2 * DINNER * 4);
    float*          usil  = (float*)         alloc((size_t)MTOK * DINNER * 4);
    unsigned short* ubf   = (unsigned short*)alloc((size_t)MTOK * DINNER * 2);
    unsigned short* xpwT  = (unsigned short*)alloc((size_t)DINNER * NXPROJ * 2);
    float*          xdbc  = (float*)         alloc((size_t)MTOK * NXPROJ * 4);
    unsigned short* dtin  = (unsigned short*)alloc((size_t)MTOK * DTRANK * 2);
    unsigned short* dtwT  = (unsigned short*)alloc((size_t)DTRANK * DINNER * 2);
    float*          dtf   = (float*)         alloc((size_t)MTOK * DINNER * 4);
    unsigned short* ybf   = (unsigned short*)alloc((size_t)MTOK * DINNER * 2);
    unsigned short* owT   = (unsigned short*)alloc((size_t)DINNER * DMODEL * 2);

    // weight transpose+convert: [K][N] f32 -> [N][K] bf16  (once per launch)
    cvt_T_bf16<<<dim3((2*DINNER)/32, DMODEL/32), 256, 0, stream>>>(w_in,  w1T,  DMODEL, 2*DINNER);
    cvt_T_bf16<<<dim3(NXPROJ/32,    DINNER/32), 256, 0, stream>>>(w_xp,  xpwT, DINNER, NXPROJ);
    cvt_T_bf16<<<dim3(DINNER/32,    DTRANK/32), 256, 0, stream>>>(w_dt,  dtwT, DTRANK, DINNER);
    cvt_T_bf16<<<dim3(DMODEL/32,    DINNER/32), 256, 0, stream>>>(w_out, owT,  DINNER, DMODEL);

    // 1) LayerNorm -> bf16
    ln_kernel<<<MTOK, 256, 0, stream>>>(x, lng, lnb, xnbf);

    // 2) xz = xn @ in_proj_w   (8192 x 4096 x 1024)
    gemm_bf16_wmma<<<dim3((2*DINNER)/64, MTOK/128), 256, 0, stream>>>(
        xnbf, w1T, xzf, nullptr, MTOK, 2*DINNER, DMODEL);

    // 3) causal depthwise conv + SiLU on u half
    conv_silu_kernel<<<(MTOK*DINNER + 255)/256, 256, 0, stream>>>(xzf, cw, cb, usil, ubf);

    // 4) xdbc = u @ x_proj_w   (8192 x 96 x 2048)
    gemm_bf16_wmma<<<dim3((NXPROJ + 63)/64, MTOK/128), 256, 0, stream>>>(
        ubf, xpwT, xdbc, nullptr, MTOK, NXPROJ, DINNER);

    // 5) dt = softplus(xdbc[:, :64] @ dt_proj_w + b)  (8192 x 2048 x 64)
    slice_dt_kernel<<<(MTOK*DTRANK + 255)/256, 256, 0, stream>>>(xdbc, dtin);
    gemm_bf16_wmma<<<dim3(DINNER/64, MTOK/128), 256, 0, stream>>>(
        dtin, dtwT, dtf, nullptr, MTOK, DINNER, DTRANK);
    softplus_kernel<<<(MTOK*DINNER + 255)/256, 256, 0, stream>>>(dtf, b_dt);

    // 6) selective scan + D skip + z-gate -> y bf16
    scan_kernel<<<(BATCH*DINNER)/256, 256, 0, stream>>>(dtf, usil, xdbc, xzf, Alog, Dp, ybf);

    // 7) out = x + y @ out_proj_w  (8192 x 1024 x 2048), fused residual
    gemm_bf16_wmma<<<dim3(DMODEL/64, MTOK/128), 256, 0, stream>>>(
        ybf, owT, out, x, MTOK, DMODEL, DINNER);
}
